// FastRCNN_33449205301967
// MI455X (gfx1250) — matrix operand order repaired and checked by reading, not compiled, not run on hardware
//
#include <hip/hip_runtime.h>
#include <hip/hip_bf16.h>

// ---------------------------------------------------------------------------
// Fast-RCNN head for MI455X (gfx1250, wave32, WMMA).
//
// roipool -> gemm_bf16x3 split-K (FC1) -> reduce(+b1,relu)
//         -> gemm_bf16x3 split-K (FC2) -> reduce(+b2,relu) -> heads
//
// FC1 streams 411 MB fp32 W1 (HBM floor ~17.6 us @ 23.3 TB/s). Weights are
// read once as fp32 and split in-register to (bf16 hi, bf16 lo); the bf16x3
// scheme (hi*hi + hi*lo + lo*hi) on v_wmma_f32_16x16x32_bf16 gives ~fp32
// accuracy at 3x bf16 issue cost (~2.7x cheaper than the fp32 16x16x4 pipe).
// Double-buffered LDS overlaps global staging with WMMA; split-K raises the
// workgroup count from 128 to 512 (FC1) / 256 (FC2) for latency hiding.
// ---------------------------------------------------------------------------

typedef __attribute__((ext_vector_type(16))) __bf16 v16bf;
typedef __attribute__((ext_vector_type(8)))  float  v8f;

union FragU {            // 32-byte WMMA operand assembled from two 16B LDS loads
    float4 q[2];
    v16bf  v;
};

#define BM 128
#define BN 128
#define BK 32

// ---------------------------------------------------------------------------
// RoIPool: x[1,512,50,50], rois[512,4] (image coords, scale 50/800 = 0.0625)
// -> pooled[512, 25088]  (flattened [C,7,7] per roi, matching reshape order)
// ---------------------------------------------------------------------------
__global__ __launch_bounds__(256) void roipool_kernel(
    const float* __restrict__ x, const float* __restrict__ rois,
    float* __restrict__ pooled)
{
    const int n = blockIdx.x;
    const float s = 0.0625f;                     // 50/800
    int x1 = (int)(rois[n * 4 + 0] * s);
    int y1 = (int)(rois[n * 4 + 1] * s);
    int x2 = (int)(rois[n * 4 + 2] * s);
    int y2 = (int)(rois[n * 4 + 3] * s);
    x1 = min(max(x1, 0), 49);  y1 = min(max(y1, 0), 49);
    x2 = min(max(x2, 0), 49);  y2 = min(max(y2, 0), 49);
    const int ww = x2 - x1 + 1;
    const int hh = y2 - y1 + 1;

    for (int idx = threadIdx.x; idx < 25088; idx += 256) {
        const int c   = idx / 49;
        const int rem = idx - c * 49;
        const int py  = rem / 7;
        const int px  = rem - py * 7;
        const int rs = y1 + (py * hh) / 7;
        const int re = y1 + ((py + 1) * hh + 6) / 7;
        const int cs = x1 + (px * ww) / 7;
        const int ce = x1 + ((px + 1) * ww + 6) / 7;
        const float* f = x + (size_t)c * 2500;
        float m = -3.402823466e38f;
        for (int h = rs; h < re; ++h)
            for (int w = cs; w < ce; ++w)
                m = fmaxf(m, f[h * 50 + w]);
        pooled[(size_t)n * 25088 + idx] = m;
    }
}

// ---------------------------------------------------------------------------
// Split-K partial GEMM: P[z][M,N] += A[M, kz..kz+Ks] @ B[kz..kz+Ks, N]
// fp32 in, fp32 partials out, bf16x3 WMMA core, double-buffered LDS.
// Block: 256 threads = 8 wave32 waves, 128x128 C tile; wave = 64x32 region
// (4x2 WMMA tiles, 3 WMMAs each per BK=32 step).
// Requires M%128==0, N%128==0, Ks%32==0.
// ---------------------------------------------------------------------------
__global__ __launch_bounds__(256) void gemm_bf16x3_splitk(
    const float* __restrict__ A, const float* __restrict__ B,
    float* __restrict__ P, int M, int N, int K, int Ks)
{
    __shared__ __align__(32) __bf16 sAh[2][BM][BK];
    __shared__ __align__(32) __bf16 sAl[2][BM][BK];
    __shared__ __align__(32) __bf16 sBh[2][BN][BK];   // transposed: [n][k]
    __shared__ __align__(32) __bf16 sBl[2][BN][BK];

    const int tid   = threadIdx.x;
    const int lane  = tid & 31;
    const int wid   = tid >> 5;
    const int waveM = wid >> 2;        // 0..1  -> 64-row half
    const int waveN = wid & 3;         // 0..3  -> 32-col quarter
    const int hs    = lane >> 4;       // half-wave select (ISA A/B layouts)
    const int lm    = lane & 15;

    const int m0   = blockIdx.y * BM;
    const int n0   = blockIdx.x * BN;
    const int kBeg = blockIdx.z * Ks;
    const int nk   = Ks / BK;

    v8f acc[4][2];
    const v8f vzero = {0.f, 0.f, 0.f, 0.f, 0.f, 0.f, 0.f, 0.f};
#pragma unroll
    for (int mt = 0; mt < 4; ++mt)
#pragma unroll
        for (int nt = 0; nt < 2; ++nt) acc[mt][nt] = vzero;

    // staging assignments: A: 2 threads/row (16 floats each); B: 8 threads/row
    const int arow = tid >> 1, acol = (tid & 1) * 16;
    const int brow = tid >> 3, bcol = (tid & 7) * 16;
    const float* aBase = A + (size_t)(m0 + arow) * K + kBeg + acol;

    float4 ra[4], rb[4];
    // ---- preload tile 0 into registers, convert+store into LDS buf 0 ------
    {
        const float4* ap = (const float4*)aBase;
        const float4* bp = (const float4*)(B + (size_t)(kBeg + brow) * N + n0 + bcol);
#pragma unroll
        for (int v = 0; v < 4; ++v) { ra[v] = ap[v]; rb[v] = bp[v]; }
#pragma unroll
        for (int v = 0; v < 4; ++v) {
            const float ea[4] = {ra[v].x, ra[v].y, ra[v].z, ra[v].w};
            const float eb[4] = {rb[v].x, rb[v].y, rb[v].z, rb[v].w};
#pragma unroll
            for (int j = 0; j < 4; ++j) {
                __bf16 h = (__bf16)ea[j];
                sAh[0][arow][acol + v * 4 + j] = h;
                sAl[0][arow][acol + v * 4 + j] = (__bf16)(ea[j] - (float)h);
                const int col = bcol + v * 4 + j;
                __bf16 g = (__bf16)eb[j];
                sBh[0][col][brow] = g;
                sBl[0][col][brow] = (__bf16)(eb[j] - (float)g);
            }
        }
    }
    __syncthreads();

    for (int kt = 0; kt < nk; ++kt) {
        const int cur  = kt & 1;
        const int nxt  = cur ^ 1;
        const bool more = (kt + 1 < nk);

        // ---- issue next tile's global loads before the WMMA block ---------
        if (more) {
            const float4* ap = (const float4*)(aBase + (kt + 1) * BK);
            const float4* bp = (const float4*)(
                B + (size_t)(kBeg + (kt + 1) * BK + brow) * N + n0 + bcol);
#pragma unroll
            for (int v = 0; v < 4; ++v) { ra[v] = ap[v]; rb[v] = bp[v]; }
        }
        if (kt + 2 < nk) {   // warm L2 for the tile after next
            __builtin_prefetch(aBase + (kt + 2) * BK, 0, 1);
            __builtin_prefetch(B + (size_t)(kBeg + (kt + 2) * BK + brow) * N + n0 + bcol, 0, 1);
        }

        // ---- gather WMMA fragments per ISA VGPR layouts -------------------
        // A (16-bit 16x32): lanes 0-15 row M=lm hold K 0..7 & 16..23,
        //                   lanes 16-31 hold K 8..15 & 24..31.
        FragU ah[4], al[4], bh[2], bl[2];
#pragma unroll
        for (int mt = 0; mt < 4; ++mt) {
            const __bf16* rh = &sAh[cur][waveM * 64 + mt * 16 + lm][0];
            const __bf16* rl = &sAl[cur][waveM * 64 + mt * 16 + lm][0];
            ah[mt].q[0] = *(const float4*)(rh + hs * 8);
            ah[mt].q[1] = *(const float4*)(rh + 16 + hs * 8);
            al[mt].q[0] = *(const float4*)(rl + hs * 8);
            al[mt].q[1] = *(const float4*)(rl + 16 + hs * 8);
        }
        // B (16-bit 32x16): lane holds column N=lm, K 0..7 & 16..23 (lanes 0-15) or
        // K 8..15 & 24..31 (lanes 16-31) -- same K order as A.
#pragma unroll
        for (int nt = 0; nt < 2; ++nt) {
            const __bf16* rh = &sBh[cur][waveN * 32 + nt * 16 + lm][0];
            const __bf16* rl = &sBl[cur][waveN * 32 + nt * 16 + lm][0];
            bh[nt].q[0] = *(const float4*)(rh + hs * 8);
            bh[nt].q[1] = *(const float4*)(rh + 16 + hs * 8);
            bl[nt].q[0] = *(const float4*)(rl + hs * 8);
            bl[nt].q[1] = *(const float4*)(rl + 16 + hs * 8);
        }

        // ---- bf16x3: hi*hi + hi*lo + lo*hi, f32 accumulate ----------------
#pragma unroll
        for (int mt = 0; mt < 4; ++mt)
#pragma unroll
            for (int nt = 0; nt < 2; ++nt) {
                acc[mt][nt] = __builtin_amdgcn_wmma_f32_16x16x32_bf16(
                    false, ah[mt].v, false, bh[nt].v, (short)0, acc[mt][nt], false, false);
                acc[mt][nt] = __builtin_amdgcn_wmma_f32_16x16x32_bf16(
                    false, ah[mt].v, false, bl[nt].v, (short)0, acc[mt][nt], false, false);
                acc[mt][nt] = __builtin_amdgcn_wmma_f32_16x16x32_bf16(
                    false, al[mt].v, false, bh[nt].v, (short)0, acc[mt][nt], false, false);
            }

        // ---- convert + store next tile into the alternate buffer ----------
        if (more) {
#pragma unroll
            for (int v = 0; v < 4; ++v) {
                const float ea[4] = {ra[v].x, ra[v].y, ra[v].z, ra[v].w};
                const float eb[4] = {rb[v].x, rb[v].y, rb[v].z, rb[v].w};
#pragma unroll
                for (int j = 0; j < 4; ++j) {
                    __bf16 h = (__bf16)ea[j];
                    sAh[nxt][arow][acol + v * 4 + j] = h;
                    sAl[nxt][arow][acol + v * 4 + j] = (__bf16)(ea[j] - (float)h);
                    const int col = bcol + v * 4 + j;
                    __bf16 g = (__bf16)eb[j];
                    sBh[nxt][col][brow] = g;
                    sBl[nxt][col][brow] = (__bf16)(eb[j] - (float)g);
                }
            }
        }
        __syncthreads();
    }

    // ---- epilogue: C/D layout — VGPR r: M = r + 8*hs, N = lane&15 ---------
    float* Cp = P + (size_t)blockIdx.z * M * N;
#pragma unroll
    for (int mt = 0; mt < 4; ++mt)
#pragma unroll
        for (int nt = 0; nt < 2; ++nt) {
            const int col = n0 + waveN * 32 + nt * 16 + lm;
#pragma unroll
            for (int r = 0; r < 8; ++r) {
                const int row = m0 + waveM * 64 + mt * 16 + r + hs * 8;
                Cp[(size_t)row * N + col] = acc[mt][nt][r];
            }
        }
}

// ---------------------------------------------------------------------------
// Deterministic split-K reduction: C[i] = relu( sum_s P[s][i] + bias[i%N] )
// ---------------------------------------------------------------------------
__global__ __launch_bounds__(256) void reduce_bias_relu(
    const float* __restrict__ P, const float* __restrict__ bias,
    float* __restrict__ C, int S, int N, size_t total, int relu)
{
    const size_t i = (size_t)blockIdx.x * 256 + threadIdx.x;
    if (i >= total) return;
    float a = 0.f;
    for (int s = 0; s < S; ++s) a += P[(size_t)s * total + i];
    a += bias[i % (size_t)N];
    if (relu) a = fmaxf(a, 0.f);
    C[i] = a;
}

// ---------------------------------------------------------------------------
// Heads: y_cls[512,21] and y_reg[512,84] from f[512,4096], written
// concatenated into d_out (cls block then reg block). Tiny-N FMA kernel.
// ---------------------------------------------------------------------------
__global__ __launch_bounds__(128) void heads_kernel(
    const float* __restrict__ f,
    const float* __restrict__ Wcls, const float* __restrict__ bcls,
    const float* __restrict__ Wreg, const float* __restrict__ breg,
    float* __restrict__ out)
{
    __shared__ float sf[4096];
    const int m = blockIdx.x;
    for (int k = threadIdx.x; k < 4096; k += 128)
        sf[k] = f[(size_t)m * 4096 + k];
    __syncthreads();

    const int j = threadIdx.x;
    if (j >= 105) return;
    float a = 0.f;
    if (j < 21) {
        for (int k = 0; k < 4096; ++k) a = fmaf(sf[k], Wcls[(size_t)k * 21 + j], a);
        out[(size_t)m * 21 + j] = a + bcls[j];
    } else {
        const int jr = j - 21;
        for (int k = 0; k < 4096; ++k) a = fmaf(sf[k], Wreg[(size_t)k * 84 + jr], a);
        out[(size_t)512 * 21 + (size_t)m * 84 + jr] = a + breg[jr];
    }
}

// ---------------------------------------------------------------------------
extern "C" void kernel_launch(void* const* d_in, const int* in_sizes, int n_in,
                              void* d_out, int out_size, void* d_ws, size_t ws_size,
                              hipStream_t stream)
{
    const float* x    = (const float*)d_in[0];   // [1,512,50,50]
    // d_in[1] = img, only shape matters (800x800) — baked into the 0.0625 scale
    const float* rois = (const float*)d_in[2];   // [512,4]
    const float* W1   = (const float*)d_in[3];   // [25088,4096]
    const float* b1   = (const float*)d_in[4];
    const float* W2   = (const float*)d_in[5];   // [4096,4096]
    const float* b2   = (const float*)d_in[6];
    const float* Wcls = (const float*)d_in[7];   // [4096,21]
    const float* bcls = (const float*)d_in[8];
    const float* Wreg = (const float*)d_in[9];   // [4096,84]
    const float* breg = (const float*)d_in[10];
    float* out = (float*)d_out;

    const int    M = 512, N = 4096;
    const size_t MN     = (size_t)M * N;                 // 2,097,152
    const size_t pooledB = (size_t)512 * 25088 * 4;      // 51.4 MB
    const size_t yB      = MN * 4;                       // 8.4 MB

    // split factors: degrade to 1 if workspace is tight (uniform per call)
    const int S1want = 4, S2want = 2;
    const bool fits = ws_size >= pooledB + (size_t)S1want * yB + 2 * yB;
    const int S1 = fits ? S1want : 1;
    const int S2 = fits ? S2want : 1;

    // workspace: pooled | partials (S1 max) | Y1 | Y2
    char* ws = (char*)d_ws;
    float* pooled = (float*)ws;
    float* part   = (float*)(ws + pooledB);
    float* Y1     = (float*)(ws + pooledB + (size_t)S1 * yB);
    float* Y2     = Y1 + MN;

    roipool_kernel<<<512, 256, 0, stream>>>(x, rois, pooled);

    // FC1: 512x25088 @ 25088x4096, split-K
    gemm_bf16x3_splitk<<<dim3(N / BN, M / BM, S1), 256, 0, stream>>>(
        pooled, W1, part, M, N, 25088, 25088 / S1);
    reduce_bias_relu<<<(int)((MN + 255) / 256), 256, 0, stream>>>(
        part, b1, Y1, S1, N, MN, 1);

    // FC2: 512x4096 @ 4096x4096, split-K
    gemm_bf16x3_splitk<<<dim3(N / BN, M / BM, S2), 256, 0, stream>>>(
        Y1, W2, part, M, N, 4096, 4096 / S2);
    reduce_bias_relu<<<(int)((MN + 255) / 256), 256, 0, stream>>>(
        part, b2, Y2, S2, N, MN, 1);

    heads_kernel<<<512, 128, 0, stream>>>(Y2, Wcls, bcls, Wreg, breg, out);
}
